// InferRNN_21680994910705
// MI455X (gfx1250) — matrix operand order repaired and checked by reading, not compiled, not run on hardware
//
#include <hip/hip_runtime.h>
#include <cmath>

typedef __attribute__((ext_vector_type(16))) _Float16 v16h;
typedef __attribute__((ext_vector_type(8)))  _Float16 v8h;
typedef __attribute__((ext_vector_type(4)))  _Float16 v4h;
typedef __attribute__((ext_vector_type(8)))  float    v8f;
typedef __attribute__((ext_vector_type(4)))  float    v4f;

constexpr int NB   = 256;   // batch
constexpr int NT   = 2048;  // time steps
constexpr int ND   = 64;    // D_IN
constexpr int NH   = 128;   // hidden
constexpr int NC   = 32;    // D_C
constexpr int MT   = 16;    // batch rows per workgroup (WMMA M)
constexpr int TCH  = 128;   // y steps buffered in LDS before flush
constexpr int HROW = NH + 8; // padded LDS row (halves) -> 272B, kills bank conflicts
constexpr int XROW = ND + 8; // padded LDS row (halves) -> 144B

// Branch-free tanh: gfx1250 V_TANH_F32 if available, else exp2/rcp fallback.
#if defined(__has_builtin)
#if __has_builtin(__builtin_amdgcn_tanhf)
#define TANH_NATIVE 1
#endif
#endif
__device__ __forceinline__ float fast_tanh(float v) {
#if defined(TANH_NATIVE)
  return __builtin_amdgcn_tanhf(v);
#else
  const float a = fminf(fabsf(v), 20.0f);
  const float e = __builtin_amdgcn_exp2f(a * 2.885390081777927f);
  const float r = 1.0f - 2.0f * __builtin_amdgcn_rcpf(e + 1.0f);
  return copysignf(r, v);
#endif
}

// Build A fragment (16x32 f16, row M = lane%16) from a row-major LDS row.
// Layout: element j -> K = half*16 + j (contiguous, matches B fragments)
__device__ __forceinline__ v16h make_a_frag(const _Float16* base, int half) {
  v8h lo = *(const v8h*)(base + half * 16);
  v8h hi = *(const v8h*)(base + half * 16 + 8);
  v16h a;
#pragma unroll
  for (int i = 0; i < 8; ++i) { a[i] = lo[i]; a[i + 8] = hi[i]; }
  return a;
}

__device__ __forceinline__ v8f wmma_f16(v16h a, v16h b, v8f c) {
  return __builtin_amdgcn_wmma_f32_16x16x32_f16(false, a, false, b, (short)0, c,
                                                false, false);
}

__global__ __launch_bounds__(256, 1) void infer_rnn_scan_wmma(
    const float* __restrict__ x,  const float* __restrict__ h0,
    const float* __restrict__ c,  const float* __restrict__ Wx,
    const float* __restrict__ Wh, const float* __restrict__ Wc,
    const float* __restrict__ bias, const float* __restrict__ Wo,
    float* __restrict__ out)
{
  __shared__ __align__(16) _Float16 hbuf[2][MT * HROW];  // recurrent state, f16, dbl-buffered
  __shared__ __align__(16) _Float16 xbuf[2][MT * XROW];  // x_t tile, f16, dbl-buffered
  __shared__ __align__(16) float    cwbuf[MT * NH];      // c@Wc + b (constant over t)
  __shared__ __align__(16) float    ypart[2][4][16];     // per-wave y partials, dbl-buffered
  __shared__ __align__(16) float    ybuf[MT * TCH];      // y staging for coalesced flush

  const int tid  = threadIdx.x;
  const int wave = tid >> 5;
  const int lane = tid & 31;
  const int half = lane >> 4;
  const int nloc = lane & 15;
  const int ncol = wave * 16 + nloc;     // this wave's H-column tile
  const int b0   = blockIdx.x * MT;      // batch rows of this workgroup

  // Rotated K-slice order for the h-chain: slice (wave&3) always sits in
  // fragment slot 0 so the y-WMMA (waves 0..3) uses a fixed register name.
  const int s0 = (wave + 0) & 3, s1 = (wave + 1) & 3;
  const int s2 = (wave + 2) & 3, s3 = (wave + 3) & 3;

  // ---------- one-time setup ----------
  // cW = c @ Wc + bias  (tiny: 16x32 @ 32x128), full fp32
  {
    const int row = tid >> 4;
    const int c0  = (tid & 15) * 8;
#pragma unroll
    for (int j = 0; j < 8; ++j) {
      const int col = c0 + j;
      float s = bias[col];
      for (int k = 0; k < NC; ++k)
        s += c[(b0 + row) * NC + k] * Wc[k * NH + col];
      cwbuf[row * NH + col] = s;
    }
  }
  // h <- h0 (f16)
#pragma unroll
  for (int i = 0; i < 8; ++i) {
    const int idx = tid * 8 + i;
    const int m = idx >> 7, col = idx & 127;
    hbuf[0][m * HROW + col] = (_Float16)h0[(b0 + m) * NH + col];
  }
  // x(t=0) tile -> LDS (f16)
  const int prow = tid >> 4;
  const int pc4  = (tid & 15) * 4;
  {
    const v4f xf = *(const v4f*)(x + ((size_t)(b0 + prow) * NT + 0) * ND + pc4);
    v4h hx = {(_Float16)xf.x, (_Float16)xf.y, (_Float16)xf.z, (_Float16)xf.w};
    *(v4h*)&xbuf[0][prow * XROW + pc4] = hx;
  }

  // Weight B-fragments, resident in VGPRs for all 2048 steps.
  // B layout (32x16): lane = column N, element j -> K = half*16 + j.
  v16h bWx[2], bWh[4], bWo;
#pragma unroll
  for (int ks = 0; ks < 2; ++ks)
#pragma unroll
    for (int j = 0; j < 16; ++j)
      bWx[ks][j] = (_Float16)Wx[(ks * 32 + half * 16 + j) * NH + ncol];
  {
    const int ss[4] = {s0, s1, s2, s3};
#pragma unroll
    for (int i = 0; i < 4; ++i)
#pragma unroll
      for (int j = 0; j < 16; ++j)
        bWh[i][j] = (_Float16)Wh[(ss[i] * 32 + half * 16 + j) * NH + ncol];
  }
  // Wo as a B fragment: column N=0 carries Wo for K-slice (wave&3), rest zero.
#pragma unroll
  for (int j = 0; j < 16; ++j)
    bWo[j] = (nloc == 0) ? (_Float16)Wo[s0 * 32 + half * 16 + j] : (_Float16)0.f;

  __syncthreads();

  // C fragment: cW tile (constant accumulator seed each step).
  v8f cfrag;
#pragma unroll
  for (int v = 0; v < 8; ++v)
    cfrag[v] = cwbuf[(v + 8 * half) * NH + ncol];

  const v8f vzero = {0.f, 0.f, 0.f, 0.f, 0.f, 0.f, 0.f, 0.f};

  // ---------- sequential scan over time ----------
#pragma unroll 2
  for (int t = 0; t < NT; ++t) {
    const int cur = t & 1, nxt = cur ^ 1;

    // Prefetch x(t+1) global -> regs (non-temporal: x is streamed exactly once).
    v4f xf = {0.f, 0.f, 0.f, 0.f};
    if (t + 1 < NT)
      xf = __builtin_nontemporal_load(
          (const v4f*)(x + ((size_t)(b0 + prow) * NT + (t + 1)) * ND + pc4));

    // Issue ALL A-fragment LDS loads up front (12 ds_load_b128 in flight).
    const _Float16* xrow = &xbuf[cur][nloc * XROW];
    const _Float16* hrow = &hbuf[cur][nloc * HROW];
    v16h a0 = make_a_frag(xrow,           half);
    v16h a1 = make_a_frag(xrow + 32,      half);
    v16h ah0 = make_a_frag(hrow + s0 * 32, half);   // slice (wave&3) first
    v16h ah1 = make_a_frag(hrow + s1 * 32, half);
    v16h ah2 = make_a_frag(hrow + s2 * 32, half);
    v16h ah3 = make_a_frag(hrow + s3 * 32, half);

    // Two independent accumulator chains (3 dependent WMMAs each).
    v8f acc0 = cfrag;
    v8f acc1 = vzero;
    acc0 = wmma_f16(a0,  bWx[0], acc0);
    acc1 = wmma_f16(a1,  bWx[1], acc1);
    acc0 = wmma_f16(ah0, bWh[0], acc0);
    acc1 = wmma_f16(ah1, bWh[1], acc1);
    acc0 = wmma_f16(ah2, bWh[2], acc0);
    acc1 = wmma_f16(ah3, bWh[3], acc1);

    // y(t-1) partial on the matrix pipe: waves 0..3 each cover one K-slice of
    // h_t @ Wo (their rotated slice s0 == wave). Replaces a 4-stage, 32x
    // ds_bpermute shuffle tree with one independent WMMA.
    if (wave < 4) {
      v8f yacc = wmma_f16(ah0, bWo, vzero);
      if (nloc == 0) {
#pragma unroll
        for (int v = 0; v < 8; ++v)
          ypart[cur][wave][v + 8 * half] = yacc[v];
      }
    }

    // h' = tanh(acc); write to next h buffer (f16).
#pragma unroll
    for (int v = 0; v < 8; ++v) {
      const float hv = fast_tanh(acc0[v] + acc1[v]);
      hbuf[nxt][(v + 8 * half) * HROW + ncol] = (_Float16)hv;
    }

    // Land prefetched x(t+1) into the other LDS buffer.
    if (t + 1 < NT) {
      v4h hx = {(_Float16)xf.x, (_Float16)xf.y, (_Float16)xf.z, (_Float16)xf.w};
      *(v4h*)&xbuf[nxt][prow * XROW + pc4] = hx;
    }

    __syncthreads();  // h', x(t+1), ypart now visible; single barrier per step

    // Combine the 4 slice partials -> y(t-1) staging slot.
    if (t > 0 && tid < 16) {
      float s = ypart[cur][0][tid] + ypart[cur][1][tid] +
                ypart[cur][2][tid] + ypart[cur][3][tid];
      ybuf[tid * TCH + ((t - 1) & (TCH - 1))] = s;
    }

    // Coalesced flush of 128 finished y steps (chunk ends at t-1).
    if ((t & (TCH - 1)) == 0 && t > 0) {
      __syncthreads();
      const int m = tid >> 4, j0 = (tid & 15) * 8;
      const int tbase = t - TCH;
      v4f y0 = *(const v4f*)&ybuf[m * TCH + j0];
      v4f y1 = *(const v4f*)&ybuf[m * TCH + j0 + 4];
      __builtin_nontemporal_store(y0, (v4f*)(out + (size_t)(b0 + m) * NT + tbase + j0));
      __builtin_nontemporal_store(y1, (v4f*)(out + (size_t)(b0 + m) * NT + tbase + j0 + 4));
      __syncthreads();
    }
  }

  // ---------- tail: y(NT-1) = h_NT @ Wo, then flush the last chunk ----------
  // Final h lives in hbuf[0] (NT even: last write went to nxt = 0).
  if (tid < 16) {
    float s = 0.f;
    for (int k = 0; k < NH; ++k)
      s += (float)hbuf[0][tid * HROW + k] * Wo[k];
    ybuf[tid * TCH + (TCH - 1)] = s;
  }
  __syncthreads();
  {
    const int m = tid >> 4, j0 = (tid & 15) * 8;
    const int tbase = NT - TCH;
    v4f y0 = *(const v4f*)&ybuf[m * TCH + j0];
    v4f y1 = *(const v4f*)&ybuf[m * TCH + j0 + 4];
    __builtin_nontemporal_store(y0, (v4f*)(out + (size_t)(b0 + m) * NT + tbase + j0));
    __builtin_nontemporal_store(y1, (v4f*)(out + (size_t)(b0 + m) * NT + tbase + j0 + 4));
  }
}

extern "C" void kernel_launch(void* const* d_in, const int* in_sizes, int n_in,
                              void* d_out, int out_size, void* d_ws, size_t ws_size,
                              hipStream_t stream) {
  (void)in_sizes; (void)n_in; (void)out_size; (void)d_ws; (void)ws_size;
  const float* x    = (const float*)d_in[0];
  const float* h0   = (const float*)d_in[1];
  const float* c    = (const float*)d_in[2];
  const float* Wx   = (const float*)d_in[3];
  const float* Wh   = (const float*)d_in[4];
  const float* Wc   = (const float*)d_in[5];
  const float* bias = (const float*)d_in[6];
  const float* Wo   = (const float*)d_in[7];
  float* out = (float*)d_out;

  dim3 grid(NB / MT);   // 16 workgroups (one 16-row batch tile each)
  dim3 block(256);      // 8 wave32s: one per 16-column tile of H=128
  hipLaunchKernelGGL(infer_rnn_scan_wmma, grid, block, 0, stream,
                     x, h0, c, Wx, Wh, Wc, bias, Wo, out);
}
